// IntegratedFluxTransformer2DModel_2731599200665
// MI455X (gfx1250) — compile-verified, matrix operand order checked by reading, not run on hardware
//
#include <hip/hip_runtime.h>
#include <hip/hip_bf16.h>

#define HS 3072
#define NH 24
#define HD 128
#define MLPD 12288
#define LIMG 2048
#define LTXT 512
#define LTOT 2560

typedef __attribute__((ext_vector_type(16))) __bf16 v16bf;
typedef __attribute__((ext_vector_type(8)))  __bf16 v8bf;
typedef __attribute__((ext_vector_type(8)))  float  v8f;
typedef __attribute__((ext_vector_type(4)))  int    v4i;

#if __has_builtin(__builtin_amdgcn_global_load_async_to_lds_b128)
#define ASYNC_LDS 1
typedef __attribute__((address_space(1))) v4i* gas_b128;
typedef __attribute__((address_space(3))) v4i* las_b128;
#else
#define ASYNC_LDS 0
#endif

__device__ inline __bf16 f2bf(float f) { return (__bf16)f; }

// WMMA fragment = two contiguous 16B chunks: elems [0..7]=p[0..7], [8..15]=p[16..23]
__device__ inline v16bf ld_frag(const __bf16* p) {
    const v8bf* q = (const v8bf*)p;
    v8bf lo = q[0];
    v8bf hi = q[2];
    return __builtin_shufflevector(lo, hi, 0,1,2,3,4,5,6,7,8,9,10,11,12,13,14,15);
}

__device__ inline v16bf cat_frag(v8bf lo, v8bf hi) {
    return __builtin_shufflevector(lo, hi, 0,1,2,3,4,5,6,7,8,9,10,11,12,13,14,15);
}

__device__ inline float gelu_tanh(float x) {
    float x3 = x * x * x;
    return 0.5f * x * (1.0f + tanhf(0.7978845608028654f * (x + 0.044715f * x3)));
}

// ---------------------------------------------------------------- silu(vec)
__global__ void silu_k(const float* __restrict__ v, float* __restrict__ o, int n) {
    int i = blockIdx.x * blockDim.x + threadIdx.x;
    if (i < n) { float x = v[i]; o[i] = x / (1.0f + __expf(-x)); }
}

// --------------------------------------------- mod vectors: silu(vec)@W + b
__global__ void modvec_k(const float* __restrict__ s, const float* __restrict__ W,
                         const float* __restrict__ b, float* __restrict__ out,
                         int N, int K) {
    int j = blockIdx.x * blockDim.x + threadIdx.x;
    if (j >= N) return;
    float acc = 0.0f;
    for (int k = 0; k < K; ++k) acc += s[k] * W[(size_t)k * N + j];
    out[j] = acc + b[j];
}

// ----------------------------------- out = shift + (1+scale)*LayerNorm(row)
__global__ __launch_bounds__(256)
void ln_mod_k(const float* __restrict__ x, const float* __restrict__ shift,
              const float* __restrict__ scl, float* __restrict__ out) {
    int row = blockIdx.x, tid = threadIdx.x;
    const float* xr = x + (size_t)row * HS;
    float s = 0.0f, ss = 0.0f;
    for (int j = tid; j < HS; j += 256) { float v = xr[j]; s += v; ss += v * v; }
    int lane = tid & 31, wid = tid >> 5;
    for (int off = 16; off; off >>= 1) { s += __shfl_xor(s, off, 32); ss += __shfl_xor(ss, off, 32); }
    __shared__ float rs[8], rss[8];
    if (lane == 0) { rs[wid] = s; rss[wid] = ss; }
    __syncthreads();
    float S = 0.0f, SS = 0.0f;
    for (int i = 0; i < 8; ++i) { S += rs[i]; SS += rss[i]; }
    float mean = S * (1.0f / HS);
    float var  = SS * (1.0f / HS) - mean * mean;
    float inv  = rsqrtf(var + 1e-6f);
    float* orow = out + (size_t)row * HS;
    for (int j = tid; j < HS; j += 256)
        orow[j] = shift[j] + (1.0f + scl[j]) * ((xr[j] - mean) * inv);
}

// ------------------------------------------------------- bf16 WMMA GEMM
// C = act(A@W + bias); optional  C = resid + gate[col]*C
// A: MxK fp32, W: KxN fp32, converted to bf16 while staging into LDS.
// A tile row-major [BM][BK]; W tile stored TRANSPOSED [BN][BK] so all
// fragment gathers are two contiguous ds_load_b128 per fragment.
#define BM 128
#define BN 128
#define BK 32
__global__ __launch_bounds__(256)
void gemm_bf16_k(const float* __restrict__ A, const float* __restrict__ W,
                 const float* __restrict__ bias, const float* __restrict__ gate,
                 const float* __restrict__ resid, float* __restrict__ Cout,
                 int M, int N, int K, int act) {
    __shared__ __attribute__((aligned(16))) __bf16 As [BM][BK];
    __shared__ __attribute__((aligned(16))) __bf16 Bst[BN][BK];   // [n][k]
    const int tid  = threadIdx.x;
    const int lane = tid & 31;
    const int wid  = tid >> 5;
    const int wr   = wid >> 1;
    const int wc   = wid & 1;
    const int m0   = blockIdx.y * BM;
    const int n0   = blockIdx.x * BN;
    const int lh   = lane >> 4;
    const int l16  = lane & 15;
    const int kb   = lh * 8;

    v8f acc[2][4];
#pragma unroll
    for (int rt = 0; rt < 2; ++rt)
#pragma unroll
        for (int ct = 0; ct < 4; ++ct)
            acc[rt][ct] = (v8f){0.f,0.f,0.f,0.f,0.f,0.f,0.f,0.f};

    for (int k0 = 0; k0 < K; k0 += BK) {
        __syncthreads();
        {   // A tile 128x32 row-major
            int r = tid >> 1, c0 = (tid & 1) * 16;
            const float4* src = (const float4*)(A + (size_t)(m0 + r) * K + k0 + c0);
            float4 f0 = src[0], f1 = src[1], f2 = src[2], f3 = src[3];
            v8bf a0, a1;
            a0[0]=f2bf(f0.x); a0[1]=f2bf(f0.y); a0[2]=f2bf(f0.z); a0[3]=f2bf(f0.w);
            a0[4]=f2bf(f1.x); a0[5]=f2bf(f1.y); a0[6]=f2bf(f1.z); a0[7]=f2bf(f1.w);
            a1[0]=f2bf(f2.x); a1[1]=f2bf(f2.y); a1[2]=f2bf(f2.z); a1[3]=f2bf(f2.w);
            a1[4]=f2bf(f3.x); a1[5]=f2bf(f3.y); a1[6]=f2bf(f3.z); a1[7]=f2bf(f3.w);
            *(v8bf*)&As[r][c0]     = a0;
            *(v8bf*)&As[r][c0 + 8] = a1;
        }
        {   // W tile 32x128 -> transposed LDS [n][k]; lane-coalesced loads
            int c = tid & 127, half = tid >> 7;
            const float* src = W + (size_t)(k0 + half * 16) * N + n0 + c;
            v8bf b0, b1;
#pragma unroll
            for (int i = 0; i < 8; ++i) b0[i] = f2bf(src[(size_t)i * N]);
#pragma unroll
            for (int i = 0; i < 8; ++i) b1[i] = f2bf(src[(size_t)(i + 8) * N]);
            *(v8bf*)&Bst[c][half * 16]     = b0;
            *(v8bf*)&Bst[c][half * 16 + 8] = b1;
            if (k0 + BK < K) __builtin_prefetch(src + (size_t)BK * N, 0, 1);
        }
        __syncthreads();

        v16bf af[2];
#pragma unroll
        for (int rt = 0; rt < 2; ++rt)
            af[rt] = ld_frag(&As[wr * 32 + rt * 16 + l16][kb]);
#pragma unroll
        for (int ct = 0; ct < 4; ++ct) {
            v16bf bf = ld_frag(&Bst[wc * 64 + ct * 16 + l16][kb]);
#pragma unroll
            for (int rt = 0; rt < 2; ++rt)
                acc[rt][ct] = __builtin_amdgcn_wmma_f32_16x16x32_bf16(
                    false, af[rt], false, bf, (short)0, acc[rt][ct], false, false);
        }
    }

#pragma unroll
    for (int rt = 0; rt < 2; ++rt)
#pragma unroll
        for (int ct = 0; ct < 4; ++ct)
#pragma unroll
            for (int e = 0; e < 8; ++e) {
                int row = m0 + wr * 32 + rt * 16 + e + lh * 8;
                int col = n0 + wc * 64 + ct * 16 + l16;
                float v = acc[rt][ct][e] + bias[col];
                if (act) v = gelu_tanh(v);
                if (gate) v = resid[(size_t)row * N + col] + gate[col] * v;
                Cout[(size_t)row * N + col] = v;
            }
}

// --------- split heads, RMS-norm q/k, RoPE, txt-first concat, emit bf16.
// softmax 1/sqrt(D) is pre-folded into q here.
__global__ __launch_bounds__(64)
void qkv_post_k(const float* __restrict__ qkv_txt, const float* __restrict__ qkv_img,
                const float* __restrict__ tqs, const float* __restrict__ tks,
                const float* __restrict__ iqs, const float* __restrict__ iks,
                const float* __restrict__ pe,
                __bf16* __restrict__ Q, __bf16* __restrict__ K, __bf16* __restrict__ V) {
    int t = blockIdx.x, h = blockIdx.y, d2 = threadIdx.x;
    const float SOFT = 0.08838834764831845f;   // 128^-0.5
    const float *src, *qsc, *ksc;
    if (t < LTXT) { src = qkv_txt + (size_t)t * (3 * HS); qsc = tqs; ksc = tks; }
    else          { src = qkv_img + (size_t)(t - LTXT) * (3 * HS); qsc = iqs; ksc = iks; }
    const float* qp = src + h * HD;
    const float* kp = src + HS + h * HD;
    const float* vp = src + 2 * HS + h * HD;
    float q0 = qp[2 * d2], q1 = qp[2 * d2 + 1];
    float k0 = kp[2 * d2], k1 = kp[2 * d2 + 1];
    float qs2 = q0 * q0 + q1 * q1, ks2 = k0 * k0 + k1 * k1;
    for (int off = 16; off; off >>= 1) { qs2 += __shfl_xor(qs2, off, 32); ks2 += __shfl_xor(ks2, off, 32); }
    __shared__ float sq[2], sk[2];
    if ((threadIdx.x & 31) == 0) { sq[threadIdx.x >> 5] = qs2; sk[threadIdx.x >> 5] = ks2; }
    __syncthreads();
    float qr = rsqrtf((sq[0] + sq[1]) * (1.0f / HD) + 1e-6f);
    float kr = rsqrtf((sk[0] + sk[1]) * (1.0f / HD) + 1e-6f);
    q0 *= qr * qsc[2 * d2]; q1 *= qr * qsc[2 * d2 + 1];
    k0 *= kr * ksc[2 * d2]; k1 *= kr * ksc[2 * d2 + 1];
    const float* p = pe + ((size_t)t * 64 + d2) * 4;   // cos,-sin,sin,cos
    float oq0 = (p[0] * q0 + p[1] * q1) * SOFT, oq1 = (p[2] * q0 + p[3] * q1) * SOFT;
    float ok0 =  p[0] * k0 + p[1] * k1,         ok1 =  p[2] * k0 + p[3] * k1;
    size_t base = ((size_t)h * LTOT + t) * HD;
    Q[base + 2 * d2] = f2bf(oq0); Q[base + 2 * d2 + 1] = f2bf(oq1);
    K[base + 2 * d2] = f2bf(ok0); K[base + 2 * d2 + 1] = f2bf(ok1);
    V[base + 2 * d2] = f2bf(vp[2 * d2]); V[base + 2 * d2 + 1] = f2bf(vp[2 * d2 + 1]);
}

// ------------------------------------------- flash attention, bf16 WMMA
// grid (LTOT/64, NH), 128 threads = 4 waves, wave owns 16 q rows.
// K/V tiles are contiguous 8KB regions in head-major bf16 layout ->
// GLOBAL_LOAD_ASYNC_TO_LDS_B128 (ASYNCcnt) when available.
// K tile row-major [key][dim]; V staged row-major then transposed in LDS
// to [dim][key] so every WMMA fragment is two contiguous ds_load_b128.
#define AKV 32
__global__ __launch_bounds__(128)
void attn_k(const __bf16* __restrict__ Q, const __bf16* __restrict__ Kb,
            const __bf16* __restrict__ Vb, float* __restrict__ O) {
    const int h   = blockIdx.y;
    const int q0  = blockIdx.x * 64;
    const int tid = threadIdx.x, lane = tid & 31, w = tid >> 5;
    const int lh = lane >> 4, l16 = lane & 15, kb = lh * 8;

    __shared__ __attribute__((aligned(16))) __bf16 Kt [AKV][HD];
    __shared__ __attribute__((aligned(16))) __bf16 Vrm[AKV][HD];
    __shared__ __attribute__((aligned(16))) __bf16 Vtt[HD][AKV];
    __shared__ __attribute__((aligned(16))) __bf16 Ps [4][16][AKV];

    const __bf16* Qh = Q  + (size_t)h * LTOT * HD;
    const __bf16* Kh = Kb + (size_t)h * LTOT * HD;
    const __bf16* Vh = Vb + (size_t)h * LTOT * HD;

    v16bf qf[4];
    {
        const v8bf* qv = (const v8bf*)(Qh + (size_t)(q0 + w * 16 + l16) * HD);
#pragma unroll
        for (int kk = 0; kk < 4; ++kk)
            qf[kk] = cat_frag(qv[kk * 4 + (kb >> 3)], qv[kk * 4 + 2 + (kb >> 3)]);
    }

    float mrow[8], lrow[8];
    v8f oacc[8];
#pragma unroll
    for (int e = 0; e < 8; ++e) { mrow[e] = -1e30f; lrow[e] = 0.0f; }
#pragma unroll
    for (int c = 0; c < 8; ++c) oacc[c] = (v8f){0.f,0.f,0.f,0.f,0.f,0.f,0.f,0.f};

    for (int kv0 = 0; kv0 < LTOT; kv0 += AKV) {
        __syncthreads();
        {   // stage K and V tiles (each 4096 bf16, contiguous in global)
            const __bf16* gk = Kh + (size_t)kv0 * HD;
            const __bf16* gv = Vh + (size_t)kv0 * HD;
            __bf16* lk = &Kt[0][0];
            __bf16* lv = &Vrm[0][0];
#if ASYNC_LDS
#pragma unroll
            for (int j = 0; j < 4; ++j) {
                int e = (tid * 4 + j) * 8;
                __builtin_amdgcn_global_load_async_to_lds_b128(
                    (gas_b128)(gk + e), (las_b128)(lk + e), 0, 0);
                __builtin_amdgcn_global_load_async_to_lds_b128(
                    (gas_b128)(gv + e), (las_b128)(lv + e), 0, 0);
            }
#if __has_builtin(__builtin_amdgcn_s_wait_asynccnt)
            __builtin_amdgcn_s_wait_asynccnt(0);
#else
            asm volatile("s_wait_asynccnt 0x0" ::: "memory");
#endif
#else
#pragma unroll
            for (int j = 0; j < 4; ++j) {
                int e = (tid * 4 + j) * 8;
                *(v8bf*)(lk + e) = *(const v8bf*)(gk + e);
                *(v8bf*)(lv + e) = *(const v8bf*)(gv + e);
            }
#endif
        }
        __syncthreads();

        {   // LDS transpose V: [key][dim] -> [dim][key]; thread owns dim c
            int c = tid;
#pragma unroll
            for (int g = 0; g < 4; ++g) {
                v8bf vv;
#pragma unroll
                for (int i = 0; i < 8; ++i) vv[i] = Vrm[g * 8 + i][c];
                *(v8bf*)&Vtt[c][g * 8] = vv;
            }
        }

        // S = q @ k^T : two 16x16 key tiles, contraction over D in 4 chunks
        v8f s0 = (v8f){0.f,0.f,0.f,0.f,0.f,0.f,0.f,0.f};
        v8f s1 = s0;
#pragma unroll
        for (int kk = 0; kk < 4; ++kk) {
            v16bf b0 = ld_frag(&Kt[l16]     [kk * 32 + kb]);
            v16bf b1 = ld_frag(&Kt[16 + l16][kk * 32 + kb]);
            s0 = __builtin_amdgcn_wmma_f32_16x16x32_bf16(false, qf[kk], false, b0, (short)0, s0, false, false);
            s1 = __builtin_amdgcn_wmma_f32_16x16x32_bf16(false, qf[kk], false, b1, (short)0, s1, false, false);
        }

        // online softmax: row stats in registers (row = e + lh*8)
        float nm[8], rescale[8];
#pragma unroll
        for (int e = 0; e < 8; ++e) {
            float a = s0[e], b = s1[e];
#pragma unroll
            for (int off = 1; off < 16; off <<= 1) {
                a = fmaxf(a, __shfl_xor(a, off, 32));
                b = fmaxf(b, __shfl_xor(b, off, 32));
            }
            nm[e] = fmaxf(mrow[e], fmaxf(a, b));
            rescale[e] = __expf(mrow[e] - nm[e]);
            mrow[e] = nm[e];
            lrow[e] *= rescale[e];
        }
#pragma unroll
        for (int c = 0; c < 8; ++c)
#pragma unroll
            for (int e = 0; e < 8; ++e) oacc[c][e] *= rescale[e];
#pragma unroll
        for (int e = 0; e < 8; ++e) {
            float p0 = __expf(s0[e] - nm[e]);
            float p1 = __expf(s1[e] - nm[e]);
            float r0 = p0, r1 = p1;
#pragma unroll
            for (int off = 1; off < 16; off <<= 1) { r0 += __shfl_xor(r0, off, 32); r1 += __shfl_xor(r1, off, 32); }
            lrow[e] += r0 + r1;
            int row = e + lh * 8;
            Ps[w][row][l16]      = f2bf(p0);
            Ps[w][row][16 + l16] = f2bf(p1);
        }
        __syncthreads();   // Ps + Vtt ready

        v16bf pf = ld_frag(&Ps[w][l16][kb]);
#pragma unroll
        for (int c = 0; c < 8; ++c) {
            v16bf vf = ld_frag(&Vtt[c * 16 + l16][kb]);
            oacc[c] = __builtin_amdgcn_wmma_f32_16x16x32_bf16(false, pf, false, vf, (short)0, oacc[c], false, false);
        }
    }

#pragma unroll
    for (int c = 0; c < 8; ++c)
#pragma unroll
        for (int e = 0; e < 8; ++e) {
            int row = q0 + w * 16 + e + lh * 8;
            int col = c * 16 + l16;
            O[(size_t)row * HS + h * HD + col] = oacc[c][e] / lrow[e];
        }
}

// ---------------------------------------------------------------- launcher
extern "C" void kernel_launch(void* const* d_in, const int* in_sizes, int n_in,
                              void* d_out, int out_size, void* d_ws, size_t ws_size,
                              hipStream_t stream) {
    const float* img = (const float*)d_in[0];
    const float* txt = (const float*)d_in[1];
    const float* vec = (const float*)d_in[2];
    const float* pe  = (const float*)d_in[3];
    const float* i_mod_w = (const float*)d_in[4];  const float* i_mod_b = (const float*)d_in[5];
    const float* i_qkv_w = (const float*)d_in[6];  const float* i_qkv_b = (const float*)d_in[7];
    const float* i_qs    = (const float*)d_in[8];  const float* i_ks    = (const float*)d_in[9];
    const float* i_pw    = (const float*)d_in[10]; const float* i_pb    = (const float*)d_in[11];
    const float* i_w1    = (const float*)d_in[12]; const float* i_b1    = (const float*)d_in[13];
    const float* i_w2    = (const float*)d_in[14]; const float* i_b2    = (const float*)d_in[15];
    const float* t_mod_w = (const float*)d_in[16]; const float* t_mod_b = (const float*)d_in[17];
    const float* t_qkv_w = (const float*)d_in[18]; const float* t_qkv_b = (const float*)d_in[19];
    const float* t_qs    = (const float*)d_in[20]; const float* t_ks    = (const float*)d_in[21];
    const float* t_pw    = (const float*)d_in[22]; const float* t_pb    = (const float*)d_in[23];
    const float* t_w1    = (const float*)d_in[24]; const float* t_b1    = (const float*)d_in[25];
    const float* t_w2    = (const float*)d_in[26]; const float* t_b2    = (const float*)d_in[27];

    float* ws = (float*)d_ws;
    const size_t o_silu = 0;
    const size_t o_imv  = 3072;
    const size_t o_tmv  = o_imv + 6 * HS;
    const size_t o_imod = o_tmv + 6 * HS;
    const size_t o_tmod = o_imod + (size_t)LIMG * HS;
    const size_t o_qkvi = o_tmod + (size_t)LTXT * HS;
    const size_t o_qkvt = o_qkvi + (size_t)LIMG * 3 * HS;
    const size_t o_q    = o_qkvt + (size_t)LTXT * 3 * HS;   // bf16 buffers (float-sized slots)
    const size_t o_k    = o_q + (size_t)NH * LTOT * HD;
    const size_t o_v    = o_k + (size_t)NH * LTOT * HD;
    const size_t o_o    = o_v + (size_t)NH * LTOT * HD;
    const size_t o_hid  = o_o + (size_t)LTOT * HS;

    __bf16* Qb = (__bf16*)(ws + o_q);
    __bf16* Kx = (__bf16*)(ws + o_k);
    __bf16* Vx = (__bf16*)(ws + o_v);

    float* dimg = (float*)d_out;
    float* dtxt = (float*)d_out + (size_t)LIMG * HS;

    silu_k<<<(HS + 255) / 256, 256, 0, stream>>>(vec, ws + o_silu, HS);
    modvec_k<<<(6 * HS) / 256, 256, 0, stream>>>(ws + o_silu, i_mod_w, i_mod_b, ws + o_imv, 6 * HS, HS);
    modvec_k<<<(6 * HS) / 256, 256, 0, stream>>>(ws + o_silu, t_mod_w, t_mod_b, ws + o_tmv, 6 * HS, HS);

    ln_mod_k<<<LIMG, 256, 0, stream>>>(img, ws + o_imv, ws + o_imv + HS, ws + o_imod);
    ln_mod_k<<<LTXT, 256, 0, stream>>>(txt, ws + o_tmv, ws + o_tmv + HS, ws + o_tmod);

    gemm_bf16_k<<<dim3(3 * HS / BN, LIMG / BM), 256, 0, stream>>>(
        ws + o_imod, i_qkv_w, i_qkv_b, nullptr, nullptr, ws + o_qkvi, LIMG, 3 * HS, HS, 0);
    gemm_bf16_k<<<dim3(3 * HS / BN, LTXT / BM), 256, 0, stream>>>(
        ws + o_tmod, t_qkv_w, t_qkv_b, nullptr, nullptr, ws + o_qkvt, LTXT, 3 * HS, HS, 0);

    qkv_post_k<<<dim3(LTOT, NH), 64, 0, stream>>>(
        ws + o_qkvt, ws + o_qkvi, t_qs, t_ks, i_qs, i_ks, pe, Qb, Kx, Vx);

    attn_k<<<dim3(LTOT / 64, NH), 128, 0, stream>>>(Qb, Kx, Vx, ws + o_o);

    gemm_bf16_k<<<dim3(HS / BN, LIMG / BM), 256, 0, stream>>>(
        ws + o_o + (size_t)LTXT * HS, i_pw, i_pb, ws + o_imv + 2 * HS, img, dimg, LIMG, HS, HS, 0);
    gemm_bf16_k<<<dim3(HS / BN, LTXT / BM), 256, 0, stream>>>(
        ws + o_o, t_pw, t_pb, ws + o_tmv + 2 * HS, txt, dtxt, LTXT, HS, HS, 0);

    ln_mod_k<<<LIMG, 256, 0, stream>>>(dimg, ws + o_imv + 3 * HS, ws + o_imv + 4 * HS, ws + o_imod);
    gemm_bf16_k<<<dim3(MLPD / BN, LIMG / BM), 256, 0, stream>>>(
        ws + o_imod, i_w1, i_b1, nullptr, nullptr, ws + o_hid, LIMG, MLPD, HS, 1);
    gemm_bf16_k<<<dim3(HS / BN, LIMG / BM), 256, 0, stream>>>(
        ws + o_hid, i_w2, i_b2, ws + o_imv + 5 * HS, dimg, dimg, LIMG, HS, MLPD, 0);

    ln_mod_k<<<LTXT, 256, 0, stream>>>(dtxt, ws + o_tmv + 3 * HS, ws + o_tmv + 4 * HS, ws + o_tmod);
    gemm_bf16_k<<<dim3(MLPD / BN, LTXT / BM), 256, 0, stream>>>(
        ws + o_tmod, t_w1, t_b1, nullptr, nullptr, ws + o_hid, LTXT, MLPD, HS, 1);
    gemm_bf16_k<<<dim3(HS / BN, LTXT / BM), 256, 0, stream>>>(
        ws + o_hid, t_w2, t_b2, ws + o_tmv + 5 * HS, dtxt, dtxt, LTXT, HS, MLPD, 0);
}
